// MultiHeadAttention_11158325035377
// MI455X (gfx1250) — matrix operand order repaired and checked by reading, not compiled, not run on hardware
//
#include <hip/hip_runtime.h>

#define DEV __device__ __forceinline__

typedef __attribute__((ext_vector_type(16))) __bf16 v16bf;
typedef __attribute__((ext_vector_type(8)))  float  v8f;
typedef __attribute__((ext_vector_type(4)))  unsigned int v4u;

union FragAB { v16bf bf; v4u u[2]; };

// Problem constants
constexpr int Bn = 8;     // batch
constexpr int Cc = 512;   // channels
constexpr int NN = 4096;  // H*W
constexpr int NH = 8;     // heads
constexpr int HD = 64;    // head dim
constexpr int O3 = 1536;  // 3*C
constexpr int KSPLIT = 8; // split-K factor for the Gram matrix

DEV unsigned short f2bf(float f) {
  unsigned u = __float_as_uint(f);
  unsigned r = (u + 0x7FFFu + ((u >> 16) & 1u)) >> 16;   // RNE
  return (unsigned short)r;
}

DEV v8f wmma_bf(v16bf a, v16bf b, v8f c) {
  return __builtin_amdgcn_wmma_f32_16x16x32_bf16(false, a, false, b, (short)0, c, false, false);
}

// A-matrix 16x32 bf16 fragment: lane half 0 -> K {0..7,16..23}, half 1 -> K {8..15,24..31}
DEV v16bf load_frag_a(const unsigned short* base, int ld, int half, int r) {
  const unsigned short* p = base + (size_t)r * ld + half * 8;
  FragAB f;
  f.u[0] = *reinterpret_cast<const v4u*>(p);
  f.u[1] = *reinterpret_cast<const v4u*>(p + 16);
  return f.bf;
}

// B-matrix 32x16 bf16 fragment from a row-major [Ncol][K] source:
// lane half 0 -> K {0..7,16..23}, half 1 -> K {8..15,24..31}, lane&15 selects column
DEV v16bf load_frag_b(const unsigned short* base, int ld, int half, int r) {
  const unsigned short* p = base + (size_t)r * ld + half * 8;
  FragAB f;
  f.u[0] = *reinterpret_cast<const v4u*>(p);
  f.u[1] = *reinterpret_cast<const v4u*>(p + 16);
  return f.bf;
}

// D-frag lane holds 8 consecutive rows at a fixed column -> 16B bf16 store
DEV void store_row8_bf(unsigned short* dst, v8f a, float addv, float scale) {
  v4u o;
#pragma unroll
  for (int p = 0; p < 4; ++p) {
    unsigned lo = f2bf((a[2 * p] + addv) * scale);
    unsigned hi = f2bf((a[2 * p + 1] + addv) * scale);
    o[p] = lo | (hi << 16);
  }
  *reinterpret_cast<v4u*>(dst) = o;
}

// ---------------- K0: weight f32 -> bf16 ----------------
__global__ void k_cvt(const float* __restrict__ s, unsigned short* __restrict__ d, int n) {
  int i = blockIdx.x * 256 + threadIdx.x;
  if (i < n) d[i] = f2bf(s[i]);
}

// ---------------- K1a: LayerNorm statistics over C, coalesced along n ----------------
__global__ __launch_bounds__(256) void k_stats(const float* __restrict__ x,
                                               float* __restrict__ mu, float* __restrict__ rs) {
  int n0 = blockIdx.x * 64, b = blockIdx.y;
  int col = threadIdx.x & 63, cg = threadIdx.x >> 6;
  const float* px = x + (size_t)b * Cc * NN + n0 + col;
  float s = 0.f, q = 0.f;
  for (int c = cg; c < Cc; c += 4) {
    float v = px[(size_t)c * NN];
    s += v; q += v * v;
  }
  __shared__ float ss[4][64], sq[4][64];
  ss[cg][col] = s; sq[cg][col] = q;
  __syncthreads();
  if (cg == 0) {
    s = ss[0][col] + ss[1][col] + ss[2][col] + ss[3][col];
    q = sq[0][col] + sq[1][col] + sq[2][col] + sq[3][col];
    float m = s * (1.f / Cc);
    float var = q * (1.f / Cc) - m * m;
    mu[b * NN + n0 + col] = m;
    rs[b * NN + n0 + col] = rsqrtf(fmaxf(var, 0.f) + 1e-5f);
  }
}

// ---------------- K1b: normalize + transpose (b,c,n)f32 -> (b,n,c)bf16 via LDS ----------------
__global__ __launch_bounds__(256) void k_norm_t(const float* __restrict__ x,
                                                const float* __restrict__ mu,
                                                const float* __restrict__ rs,
                                                unsigned short* __restrict__ xn) {
  __shared__ unsigned short tile[32][33];
  int n0 = blockIdx.x * 32, c0 = blockIdx.y * 32, b = blockIdx.z;
  int ln = threadIdx.x & 31, rg = threadIdx.x >> 5;  // 8 row-groups
  float m = mu[b * NN + n0 + ln], rr = rs[b * NN + n0 + ln];
#pragma unroll
  for (int i = 0; i < 4; ++i) {
    int c = c0 + rg + i * 8;
    float v = x[(size_t)(b * Cc + c) * NN + n0 + ln];
    tile[rg + i * 8][ln] = f2bf((v - m) * rr);
  }
  __syncthreads();
  // write phase: pack two bf16 per lane -> 4B coalesced stores along c
  int c2 = threadIdx.x & 15;        // 16 c-pairs = 32 channels
  int nr0 = threadIdx.x >> 4;       // 16 n rows, two passes
#pragma unroll
  for (int i = 0; i < 2; ++i) {
    int nr = nr0 + i * 16;
    unsigned lo = tile[2 * c2][nr];
    unsigned hi = tile[2 * c2 + 1][nr];
    *reinterpret_cast<unsigned*>(&xn[(size_t)(b * NN + n0 + nr) * Cc + c0 + 2 * c2]) =
        lo | (hi << 16);
  }
}

// ---------------- K2: QKV GEMM (M=n 4096, N=o 1536, K=c 512), LDS-staged ----------------
// Epilogue scatters: q -> qT[b][h][d][n] (x0.125), k -> kT[b][h][d][n], v -> vb[b][n][c]
constexpr int LDP = 56;  // LDS row stride in halves (112B: 16B-aligned, conflict-free b128 reads)

__global__ __launch_bounds__(256) void k_qkv(const unsigned short* __restrict__ xn,
                                             const unsigned short* __restrict__ wq,
                                             const float* __restrict__ qkvb,
                                             unsigned short* __restrict__ qT,
                                             unsigned short* __restrict__ kT,
                                             unsigned short* __restrict__ vb) {
  __shared__ unsigned short sA[128 * LDP];
  __shared__ unsigned short sB[128 * LDP];
  int o0 = blockIdx.x * 128, n0 = blockIdx.y * 128, b = blockIdx.z;
  int tid = threadIdx.x, wave = tid >> 5, lane = tid & 31;
  int wm = wave >> 2, wn = wave & 3;
  int half = lane >> 4, r = lane & 15;
  const unsigned short* gA = xn + (size_t)(b * NN + n0) * Cc;
  const unsigned short* gB = wq + (size_t)o0 * Cc;
  v8f acc[4][2] = {};
  for (int k0 = 0; k0 < Cc; k0 += 32) {
#pragma unroll
    for (int rep = 0; rep < 2; ++rep) {
      int slot = tid + rep * 256;          // 512 slots = 128 rows x 4
      int row = slot >> 2;
      int ko = (slot & 3) * 8;
      *reinterpret_cast<v4u*>(&sA[row * LDP + ko]) =
          *reinterpret_cast<const v4u*>(gA + (size_t)row * Cc + k0 + ko);
      *reinterpret_cast<v4u*>(&sB[row * LDP + ko]) =
          *reinterpret_cast<const v4u*>(gB + (size_t)row * Cc + k0 + ko);
    }
    if (k0 + 32 < Cc) {                    // global_prefetch_b8 of next K tile
      __builtin_prefetch(gA + (size_t)(tid >> 1) * Cc + k0 + 32, 0, 1);
      __builtin_prefetch(gB + (size_t)(tid >> 1) * Cc + k0 + 32, 0, 1);
    }
    __syncthreads();
    FragAB a[4], bb[2];
#pragma unroll
    for (int mi = 0; mi < 4; ++mi)
      a[mi].bf = load_frag_a(&sA[(wm * 64 + mi * 16) * LDP], LDP, half, r);
#pragma unroll
    for (int ni = 0; ni < 2; ++ni)
      bb[ni].bf = load_frag_b(&sB[(wn * 32 + ni * 16) * LDP], LDP, half, r);
#pragma unroll
    for (int mi = 0; mi < 4; ++mi)
#pragma unroll
      for (int ni = 0; ni < 2; ++ni)
        acc[mi][ni] = wmma_bf(a[mi].bf, bb[ni].bf, acc[mi][ni]);
    __syncthreads();
  }
#pragma unroll
  for (int ni = 0; ni < 2; ++ni) {
    int o = o0 + wn * 32 + ni * 16 + r;    // column (fixed per lane)
    float bv = qkvb[o];
    int t = o >> 9, oc = o & 511;          // q/k/v section, channel within
#pragma unroll
    for (int mi = 0; mi < 4; ++mi) {
      int nb = n0 + wm * 64 + mi * 16 + half * 8;  // 8 consecutive n rows
      if (t == 2) {
#pragma unroll
        for (int e = 0; e < 8; ++e)
          vb[(size_t)(b * NN + nb + e) * Cc + oc] = f2bf(acc[mi][ni][e] + bv);
      } else {
        unsigned short* dst = (t == 0 ? qT : kT) + (size_t)(b * Cc + oc) * NN + nb;
        store_row8_bf(dst, acc[mi][ni], bv, t == 0 ? 0.125f : 1.0f);
      }
    }
  }
}

// ---------------- K3a: split-K partial Gram matrix Q^T K (64x64, K slice 512) ----------------
__global__ __launch_bounds__(256) void k_attn_part(const unsigned short* __restrict__ qT,
                                                   const unsigned short* __restrict__ kT,
                                                   float* __restrict__ part) {
  int split = blockIdx.x;                   // 0..KSPLIT-1
  int bh = blockIdx.y;
  int tid = threadIdx.x, wave = tid >> 5, lane = tid & 31;
  int half = lane >> 4, r = lane & 15;
  const int KLEN = NN / KSPLIT;             // 512
  const unsigned short* qb = qT + (size_t)bh * HD * NN + split * KLEN;
  const unsigned short* kb = kT + (size_t)bh * HD * NN + split * KLEN;
  int ti = wave >> 1;                       // wave handles tiles (ti,tj0),(ti,tj0+1)
  int tj0 = (wave & 1) * 2;
  v8f acc0 = {}, acc1 = {};
  for (int k = 0; k < KLEN; k += 32) {
    v16bf a = load_frag_a(qb + (size_t)(ti * 16) * NN + k, NN, half, r);
    v16bf b0 = load_frag_b(kb + (size_t)(tj0 * 16) * NN + k, NN, half, r);
    v16bf b1 = load_frag_b(kb + (size_t)((tj0 + 1) * 16) * NN + k, NN, half, r);
    acc0 = wmma_bf(a, b0, acc0);
    acc1 = wmma_bf(a, b1, acc1);
  }
  float* dst = part + ((size_t)bh * KSPLIT + split) * 4096;
#pragma unroll
  for (int e = 0; e < 8; ++e) {
    int row = ti * 16 + half * 8 + e;
    dst[row * 64 + tj0 * 16 + r]       = acc0[e];
    dst[row * 64 + (tj0 + 1) * 16 + r] = acc1[e];
  }
}

// ---------------- K3b: reduce partials + clip + softmax -> bf16 attn ----------------
__global__ __launch_bounds__(256) void k_attn_red(const float* __restrict__ part,
                                                  unsigned short* __restrict__ attnb) {
  __shared__ float sAt[64 * 66];
  int bh = blockIdx.x;
  const float* p = part + (size_t)bh * KSPLIT * 4096;
  for (int i = threadIdx.x; i < 4096; i += 256) {
    float s = 0.f;
#pragma unroll
    for (int sp = 0; sp < KSPLIT; ++sp) s += p[sp * 4096 + i];
    int row = i >> 6, col = i & 63;
    sAt[row * 66 + col] = fminf(fmaxf(s, -50.f), 50.f);
  }
  __syncthreads();
  if (threadIdx.x < 64) {
    int row = threadIdx.x;
    float m = -3.4e38f;
    for (int j = 0; j < 64; ++j) m = fmaxf(m, sAt[row * 66 + j]);
    float s = 0.f;
    for (int j = 0; j < 64; ++j) s += __expf(sAt[row * 66 + j] - m);
    float inv = 1.f / s;
    for (int j = 0; j < 64; ++j) {
      float pr = __expf(sAt[row * 66 + j] - m) * inv;
      if (pr != pr) pr = 1.0f / 64.0f;      // NaN guard per reference
      attnb[((size_t)bh * 64 + row) * 64 + j] = f2bf(pr);
    }
  }
}

// ---------------- K4: out = attn(64x64) x V^T, per (b,h); writes out_t[b][n][c] bf16 ----------------
__global__ __launch_bounds__(256) void k_out(const unsigned short* __restrict__ attnb,
                                             const unsigned short* __restrict__ vb,
                                             unsigned short* __restrict__ out_t) {
  int n0 = blockIdx.x * 128, bh = blockIdx.y;
  int b = bh >> 3, h = bh & 7;
  int tid = threadIdx.x, wave = tid >> 5, lane = tid & 31;
  int half = lane >> 4, r = lane & 15;
  const unsigned short* ab = attnb + (size_t)bh * 64 * 64;
  int nbase = n0 + wave * 16;               // each wave owns 16 spatial columns
  v8f acc[4] = {};
  for (int k = 0; k < 64; k += 32) {
    v16bf bfrag = load_frag_b(vb + (size_t)(b * NN + nbase) * Cc + h * 64 + k, Cc, half, r);
#pragma unroll
    for (int mi = 0; mi < 4; ++mi) {
      v16bf a = load_frag_a(ab + (size_t)(mi * 16) * 64 + k, 64, half, r);
      acc[mi] = wmma_bf(a, bfrag, acc[mi]);
    }
  }
  int n = nbase + r;
#pragma unroll
  for (int mi = 0; mi < 4; ++mi) {
    unsigned short* dst = out_t + (size_t)(b * NN + n) * Cc + h * 64 + mi * 16 + half * 8;
    store_row8_bf(dst, acc[mi], 0.f, 1.f);  // transposed store is contiguous in c
  }
}

// ---------------- K5: proj GEMM y[b][o][n] = proj_w(512x512) * out_t + bias, f32 out ----------------
__global__ __launch_bounds__(256) void k_proj(const unsigned short* __restrict__ wp,
                                              const unsigned short* __restrict__ out_t,
                                              const float* __restrict__ pb,
                                              float* __restrict__ y) {
  __shared__ unsigned short sA[128 * LDP];
  __shared__ unsigned short sB[128 * LDP];
  int o0 = blockIdx.x * 128, n0 = blockIdx.y * 128, b = blockIdx.z;
  int tid = threadIdx.x, wave = tid >> 5, lane = tid & 31;
  int wm = wave >> 2, wn = wave & 3;
  int half = lane >> 4, r = lane & 15;
  const unsigned short* gA = wp + (size_t)o0 * Cc;
  const unsigned short* gB = out_t + (size_t)(b * NN + n0) * Cc;
  v8f acc[4][2] = {};
  for (int k0 = 0; k0 < Cc; k0 += 32) {
#pragma unroll
    for (int rep = 0; rep < 2; ++rep) {
      int slot = tid + rep * 256;
      int row = slot >> 2;
      int ko = (slot & 3) * 8;
      *reinterpret_cast<v4u*>(&sA[row * LDP + ko]) =
          *reinterpret_cast<const v4u*>(gA + (size_t)row * Cc + k0 + ko);
      *reinterpret_cast<v4u*>(&sB[row * LDP + ko]) =
          *reinterpret_cast<const v4u*>(gB + (size_t)row * Cc + k0 + ko);
    }
    if (k0 + 32 < Cc) {
      __builtin_prefetch(gB + (size_t)(tid >> 1) * Cc + k0 + 32, 0, 1);
    }
    __syncthreads();
    FragAB a[4], bb[2];
#pragma unroll
    for (int mi = 0; mi < 4; ++mi)
      a[mi].bf = load_frag_a(&sA[(wm * 64 + mi * 16) * LDP], LDP, half, r);
#pragma unroll
    for (int ni = 0; ni < 2; ++ni)
      bb[ni].bf = load_frag_b(&sB[(wn * 32 + ni * 16) * LDP], LDP, half, r);
#pragma unroll
    for (int mi = 0; mi < 4; ++mi)
#pragma unroll
      for (int ni = 0; ni < 2; ++ni)
        acc[mi][ni] = wmma_bf(a[mi].bf, bb[ni].bf, acc[mi][ni]);
    __syncthreads();
  }
#pragma unroll
  for (int mi = 0; mi < 4; ++mi) {
    int ob = o0 + wm * 64 + mi * 16 + half * 8;   // 8 consecutive output channels
#pragma unroll
    for (int ni = 0; ni < 2; ++ni) {
      int n = n0 + wn * 32 + ni * 16 + r;
#pragma unroll
      for (int e = 0; e < 8; ++e)
        y[(size_t)(b * Cc + ob + e) * NN + n] = acc[mi][ni][e] + pb[ob + e];
    }
  }
}

extern "C" void kernel_launch(void* const* d_in, const int* in_sizes, int n_in,
                              void* d_out, int out_size, void* d_ws, size_t ws_size,
                              hipStream_t stream) {
  const float* x      = (const float*)d_in[0];
  const float* qkv_w  = (const float*)d_in[1];
  const float* qkv_b  = (const float*)d_in[2];
  const float* proj_w = (const float*)d_in[3];
  const float* proj_b = (const float*)d_in[4];
  float* y = (float*)d_out;

  char* ws = (char*)d_ws;
  const size_t SZ = (size_t)Bn * NN * Cc * 2;        // 32 MB per bf16 tensor
  unsigned short* xn = (unsigned short*)(ws);        // reused as out_t after K2
  unsigned short* qT = (unsigned short*)(ws + SZ);
  unsigned short* kT = (unsigned short*)(ws + 2 * SZ);
  unsigned short* vb = (unsigned short*)(ws + 3 * SZ);
  size_t off = 4 * SZ;
  float* mu = (float*)(ws + off);
  float* rs = mu + Bn * NN;                 off += (size_t)Bn * NN * 2 * 4;
  unsigned short* wq = (unsigned short*)(ws + off);    off += (size_t)O3 * Cc * 2;
  unsigned short* wp = (unsigned short*)(ws + off);    off += (size_t)Cc * Cc * 2;
  unsigned short* attnb = (unsigned short*)(ws + off); off += (size_t)Bn * NH * 64 * 64 * 2;
  float* part = (float*)(ws + off);         // Bn*NH*KSPLIT*64*64*4 = 8 MB
  unsigned short* out_t = xn;

  k_cvt<<<(O3 * Cc + 255) / 256, 256, 0, stream>>>(qkv_w, wq, O3 * Cc);
  k_cvt<<<(Cc * Cc + 255) / 256, 256, 0, stream>>>(proj_w, wp, Cc * Cc);
  k_stats<<<dim3(NN / 64, Bn), 256, 0, stream>>>(x, mu, rs);
  k_norm_t<<<dim3(NN / 32, Cc / 32, Bn), 256, 0, stream>>>(x, mu, rs, xn);
  k_qkv<<<dim3(O3 / 128, NN / 128, Bn), 256, 0, stream>>>(xn, wq, qkv_b, qT, kT, vb);
  k_attn_part<<<dim3(KSPLIT, Bn * NH), 256, 0, stream>>>(qT, kT, part);
  k_attn_red<<<dim3(Bn * NH), 256, 0, stream>>>(part, attnb);
  k_out<<<dim3(NN / 128, Bn * NH), 256, 0, stream>>>(attnb, vb, out_t);
  k_proj<<<dim3(Cc / 128, NN / 128, Bn), 256, 0, stream>>>(wp, out_t, proj_b, y);
}
